// Encoder_81509889343552
// MI455X (gfx1250) — matrix operand order repaired and checked by reading, not compiled, not run on hardware
//
#include <hip/hip_runtime.h>
#include <hip/hip_bf16.h>

// ---------------------------------------------------------------------------
// Types for WMMA fragments (gfx1250, wave32)
// ---------------------------------------------------------------------------
typedef __attribute__((ext_vector_type(16))) _Float16 v16h;
typedef __attribute__((ext_vector_type(8)))  _Float16 v8h;
typedef __attribute__((ext_vector_type(8)))  float    v8f;

#define NB   4          // batch
#define HH   512
#define WW   512
#define NS   64         // stat slices
#define EPSV 1e-5f

// ---------------------------------------------------------------------------
// Workspace layout (bytes)
// ---------------------------------------------------------------------------
static constexpr size_t SZ_RAW  = (size_t)NB * HH * WW * 64 * 4;        // 256 MB f32 raw conv out
static constexpr size_t SZ_PAD  = (size_t)NB * 514 * 514 * 128 * 2;    // ~270 MB f16 padded act
static constexpr size_t SZ_BP   = (size_t)9216 * 512 * 2;              // 9.4 MB packed weights
static constexpr size_t SZ_PS   = (size_t)NS * NB * 1024 * 4;          // 1 MB partial sums
static constexpr size_t OFF_RAW  = 0;
static constexpr size_t OFF_PADA = OFF_RAW  + SZ_RAW;
static constexpr size_t OFF_PADB = OFF_PADA + SZ_PAD;
static constexpr size_t OFF_BP   = OFF_PADB + SZ_PAD;
static constexpr size_t OFF_PSUM = OFF_BP   + SZ_BP;
static constexpr size_t OFF_PSQ  = OFF_PSUM + SZ_PS;
static constexpr size_t OFF_MEAN = OFF_PSQ  + SZ_PS;
static constexpr size_t OFF_RSTD = OFF_MEAN + (size_t)NB * 1024 * 4;
static constexpr size_t OFF_POOL = OFF_RSTD + (size_t)NB * 1024 * 4;   // 12 acc + 4 cnt floats

__device__ __forceinline__ int reflect_idx(int i, int n) {
  if (i < 0) i = -i;
  if (i >= n) i = 2 * n - 2 - i;
  return i;
}

// ---------------------------------------------------------------------------
// Stem: reflect-pad(3) 7x7 conv, 3 -> 64, NCHW f32 in, NHWC f32 out (no bias:
// bias cancels in the following InstanceNorm).
// ---------------------------------------------------------------------------
__global__ void conv0_kernel(const float* __restrict__ x, const float* __restrict__ w0,
                             float* __restrict__ out) {
  int id = blockIdx.x * blockDim.x + threadIdx.x;      // B*H*W*64 threads
  int co = id & 63;
  int p  = id >> 6;
  int ox = p % WW;
  int t  = p / WW;
  int oy = t % HH;
  int b  = t / HH;
  float acc = 0.f;
  for (int ky = 0; ky < 7; ++ky) {
    int iy = reflect_idx(oy + ky - 3, HH);
    for (int kx = 0; kx < 7; ++kx) {
      int ix = reflect_idx(ox + kx - 3, WW);
      #pragma unroll
      for (int ci = 0; ci < 3; ++ci) {
        float v = x[(((size_t)b * 3 + ci) * HH + iy) * WW + ix];
        float w = w0[(((size_t)co * 3 + ci) * 7 + ky) * 7 + kx];
        acc = fmaf(v, w, acc);
      }
    }
  }
  out[(size_t)p * 64 + co] = acc;
}

// ---------------------------------------------------------------------------
// Instance-norm statistics: sliced partial sums (coalesced over channels)
// raw layout (B, P, C) f32.  psum/psq: [s][b][c]
// ---------------------------------------------------------------------------
__global__ void stats_partial_kernel(const float* __restrict__ raw,
                                     float* __restrict__ psum, float* __restrict__ psq,
                                     int P, int C) {
  int s = blockIdx.x;   // slice
  int b = blockIdx.y;
  for (int c = threadIdx.x; c < C; c += blockDim.x) {
    float sum = 0.f, sq = 0.f;
    for (int p = s; p < P; p += NS) {
      float v = raw[((size_t)b * P + p) * C + c];
      sum += v;
      sq  = fmaf(v, v, sq);
    }
    psum[((size_t)s * NB + b) * C + c] = sum;
    psq [((size_t)s * NB + b) * C + c] = sq;
  }
}

__global__ void stats_combine_kernel(const float* __restrict__ psum, const float* __restrict__ psq,
                                     float* __restrict__ mean, float* __restrict__ rstd,
                                     int P, int C) {
  int b = blockIdx.x;
  for (int c = threadIdx.x; c < C; c += blockDim.x) {
    float sum = 0.f, sq = 0.f;
    for (int s = 0; s < NS; ++s) {
      sum += psum[((size_t)s * NB + b) * C + c];
      sq  += psq [((size_t)s * NB + b) * C + c];
    }
    float m = sum / (float)P;
    float v = sq / (float)P - m * m;           // biased var (torch InstanceNorm2d)
    mean[b * C + c] = m;
    rstd[b * C + c] = rsqrtf(v + EPSV);
  }
}

// ---------------------------------------------------------------------------
// Normalize + ReLU + f16 pack into next-layer input buffer.
// mode 0: zero-pad 1 (Hp=Hin+2)            -- input to stride-2 conv
// mode 1: zero-dilate x2 + pad(lo1,hi2)    -- input to "transposed" conv
// mode 2: plain (Hp=Hin)                   -- input to final 7x7 head
// ---------------------------------------------------------------------------
__global__ void norm_kernel(const float* __restrict__ raw,
                            const float* __restrict__ mean, const float* __restrict__ rstd,
                            _Float16* __restrict__ dst,
                            int Hin, int Win, int C, int Hp, int Wp, int mode, size_t total) {
  size_t id = (size_t)blockIdx.x * blockDim.x + threadIdx.x;
  if (id >= total) return;
  int c = (int)(id % C);
  size_t q = id / C;
  int xx = (int)(q % Wp); q /= Wp;
  int yy = (int)(q % Hp);
  int b  = (int)(q / Hp);
  float val = 0.f;
  int iy, ix;
  bool valid;
  if (mode == 0) {
    iy = yy - 1; ix = xx - 1;
    valid = (iy >= 0) && (iy < Hin) && (ix >= 0) && (ix < Win);
  } else if (mode == 1) {
    valid = (yy & 1) && (xx & 1);
    iy = (yy - 1) >> 1; ix = (xx - 1) >> 1;
    valid = valid && (iy < Hin) && (ix < Win);
  } else {
    iy = yy; ix = xx; valid = true;
  }
  if (valid) {
    float v = raw[(((size_t)b * Hin + iy) * Win + ix) * C + c];
    v = (v - mean[b * C + c]) * rstd[b * C + c];
    val = v > 0.f ? v : 0.f;
  }
  dst[id] = (_Float16)val;
}

// ---------------------------------------------------------------------------
// Weight packing into WMMA B-fragment order.
// Linear index = ((kb*(N/16) + nb)*32 + lane)*16 + j ; K index = kb*32 + (j&7) + (lane>=16?8:0) + (j>=8?16:0)
// K ordering = (ky*3+kx)*Cin + ci  (so a 32-wide K block stays in one tap, Cin%32==0)
// ---------------------------------------------------------------------------
__global__ void pack_down_kernel(const float* __restrict__ w /*[N][Cin][3][3]*/,
                                 _Float16* __restrict__ bp, int Cin, int N, int totalKN) {
  int id = blockIdx.x * blockDim.x + threadIdx.x;
  if (id >= totalKN) return;
  int j    = id & 15;
  int lane = (id >> 4) & 31;
  int rest = id >> 9;
  int nbt  = N >> 4;
  int nb   = rest % nbt;
  int kb   = rest / nbt;
  int n = (nb << 4) + (lane & 15);
  int k = (kb << 5) + (j & 7) + ((lane >> 4) << 3) + ((j >> 3) << 4);
  int t  = k / Cin;
  int ci = k - t * Cin;
  int ky = t / 3;
  int kx = t - ky * 3;
  bp[id] = (_Float16)w[(((size_t)n * Cin + ci) * 3 + ky) * 3 + kx];
}

__global__ void pack_up_kernel(const float* __restrict__ w /*[Cin][N][3][3] torch ConvT*/,
                               _Float16* __restrict__ bp, int Cin, int N, int totalKN) {
  int id = blockIdx.x * blockDim.x + threadIdx.x;
  if (id >= totalKN) return;
  int j    = id & 15;
  int lane = (id >> 4) & 31;
  int rest = id >> 9;
  int nbt  = N >> 4;
  int nb   = rest % nbt;
  int kb   = rest / nbt;
  int n = (nb << 4) + (lane & 15);
  int k = (kb << 5) + (j & 7) + ((lane >> 4) << 3) + ((j >> 3) << 4);
  int t  = k / Cin;
  int ci = k - t * Cin;
  int ky = t / 3;
  int kx = t - ky * 3;
  // flipped kernel (transposed conv as conv on dilated input)
  bp[id] = (_Float16)w[(((size_t)ci * N + n) * 3 + (2 - ky)) * 3 + (2 - kx)];
}

// ---------------------------------------------------------------------------
// Implicit-GEMM 3x3 conv via V_WMMA_F32_16X16X32_F16, templated on Cin.
// inp: (B, Hp, Wp, C) f16 (pre-padded / pre-dilated), valid conv, stride s.
// bp:  fragment-packed weights (K=9C by N).
// out: (B*Ho*Wo, N) f32.
// One wave = 16 output pixels x 64 output channels (4 accumulators).
// K loop: incremental tap walker (no divisions) + ping-pong double buffering
// unrolled by 2 so B/A loads overlap the previous block's WMMAs.
// ---------------------------------------------------------------------------
template <int C>
__global__ __launch_bounds__(128) void gemm_conv3x3_kernel(
    const _Float16* __restrict__ inp, const _Float16* __restrict__ bp,
    float* __restrict__ outp, int Hp, int Wp, int N,
    int Ho, int Wo, int stride) {
  constexpr int KB = (9 * C) >> 5;   // K blocks of 32 (always even for C>=64)
  constexpr int CB = C >> 5;         // K blocks per tap
  const int lane = threadIdx.x & 31;
  const int wave = threadIdx.x >> 5;
  const int half = lane >> 4;
  const int row  = lane & 15;
  const int m0 = (blockIdx.x * 4 + wave) * 16;
  const int n0 = blockIdx.y * 64;
  const int HW = Ho * Wo;

  // Per-lane A row (M = lane&15): decode output pixel -> input base pointer
  int m  = m0 + row;
  int b  = m / HW;
  int r2 = m - b * HW;
  int oy = r2 / Wo;
  int ox = r2 - oy * Wo;
  const _Float16* abase =
      inp + ((size_t)(b * Hp + oy * stride) * Wp + ox * stride) * C + half * 8;

  const int    nbt   = N >> 4;
  const size_t bstep = (size_t)nbt * 512;          // packed-B stride per K block
  const _Float16* bptr = bp + ((size_t)(n0 >> 4) * 32 + lane) * 16;

  // Incremental tap walker state (no divisions in the loop)
  const _Float16* atap = abase;                    // current tap base
  int cb = 0;                                      // channel block within tap
  int kx = 0;                                      // tap x within row
  const size_t rowstep = (size_t)Wp * C - 3 * (size_t)C;

  auto loadA = [&](v16h& a) {
    const _Float16* ap = atap + (cb << 5);
    v8h lo = *(const v8h*)ap;          // K offs  0.. 7 (+8 for upper lanes)
    v8h hi = *(const v8h*)(ap + 16);   // K offs 16..23 (+8 for upper lanes)
    #pragma unroll
    for (int j = 0; j < 8; ++j) { a[j] = lo[j]; a[j + 8] = hi[j]; }
  };
  auto loadB = [&](v16h& b0, v16h& b1, v16h& b2, v16h& b3) {
    b0 = *(const v16h*)(bptr);
    b1 = *(const v16h*)(bptr + 512);
    b2 = *(const v16h*)(bptr + 1024);
    b3 = *(const v16h*)(bptr + 1536);
  };
  auto advance = [&]() {
    if (++cb == CB) {
      cb = 0;
      atap += C;                        // next tap in row
      if (++kx == 3) { kx = 0; atap += rowstep; }
    }
    bptr += bstep;
  };

  v8f acc0 = {}, acc1 = {}, acc2 = {}, acc3 = {};
  v16h aX, aY, bX0, bX1, bX2, bX3, bY0, bY1, bY2, bY3;

  // Prologue: fetch block 0 into set X
  loadA(aX);
  loadB(bX0, bX1, bX2, bX3);
  advance();

  for (int kb = 0; kb < KB; kb += 2) {
    // Issue loads for block kb+1 (always exists: KB even) before consuming X.
    __builtin_prefetch(bptr + bstep, 0, 0);
    loadA(aY);
    loadB(bY0, bY1, bY2, bY3);
    advance();

    acc0 = __builtin_amdgcn_wmma_f32_16x16x32_f16(false, aX, false, bX0, (short)0, acc0, false, false);
    acc1 = __builtin_amdgcn_wmma_f32_16x16x32_f16(false, aX, false, bX1, (short)0, acc1, false, false);
    acc2 = __builtin_amdgcn_wmma_f32_16x16x32_f16(false, aX, false, bX2, (short)0, acc2, false, false);
    acc3 = __builtin_amdgcn_wmma_f32_16x16x32_f16(false, aX, false, bX3, (short)0, acc3, false, false);

    if (kb + 2 < KB) {                 // loads for block kb+2 overlap Y's WMMAs
      loadA(aX);
      loadB(bX0, bX1, bX2, bX3);
      advance();
    }

    acc0 = __builtin_amdgcn_wmma_f32_16x16x32_f16(false, aY, false, bY0, (short)0, acc0, false, false);
    acc1 = __builtin_amdgcn_wmma_f32_16x16x32_f16(false, aY, false, bY1, (short)0, acc1, false, false);
    acc2 = __builtin_amdgcn_wmma_f32_16x16x32_f16(false, aY, false, bY2, (short)0, acc2, false, false);
    acc3 = __builtin_amdgcn_wmma_f32_16x16x32_f16(false, aY, false, bY3, (short)0, acc3, false, false);
  }

  // C/D layout: elem r -> (M = r + half*8, N = n0 + nb*16 + row)
  float* orow = outp + (size_t)(m0 + half * 8) * N + n0 + row;
  #pragma unroll
  for (int r = 0; r < 8; ++r) {
    float* o = orow + (size_t)r * N;
    o[0]  = acc0[r];
    o[16] = acc1[r];
    o[32] = acc2[r];
    o[48] = acc3[r];
  }
}

// ---------------------------------------------------------------------------
// Head: reflect-pad(3) 7x7 conv 64 -> 3 (+bias) + tanh.
// act: (B,H,W,64) f16 ; h: (B,H*W,3) f32
// ---------------------------------------------------------------------------
__global__ void convf_kernel(const _Float16* __restrict__ act, const float* __restrict__ wf,
                             const float* __restrict__ bf, float* __restrict__ h) {
  int p = blockIdx.x * blockDim.x + threadIdx.x;   // B*H*W threads
  int ox = p % WW;
  int t  = p / WW;
  int oy = t % HH;
  int b  = t / HH;
  float a0 = 0.f, a1 = 0.f, a2 = 0.f;
  for (int ky = 0; ky < 7; ++ky) {
    int iy = reflect_idx(oy + ky - 3, HH);
    for (int kx = 0; kx < 7; ++kx) {
      int ix = reflect_idx(ox + kx - 3, WW);
      const _Float16* ap = act + (((size_t)b * HH + iy) * WW + ix) * 64;
      for (int ci = 0; ci < 64; ++ci) {
        float v = (float)ap[ci];
        a0 = fmaf(v, wf[(((size_t)0 * 64 + ci) * 7 + ky) * 7 + kx], a0);
        a1 = fmaf(v, wf[(((size_t)1 * 64 + ci) * 7 + ky) * 7 + kx], a1);
        a2 = fmaf(v, wf[(((size_t)2 * 64 + ci) * 7 + ky) * 7 + kx], a2);
      }
    }
  }
  float* hp = h + (size_t)p * 3;
  hp[0] = tanhf(a0 + bf[0]);
  hp[1] = tanhf(a1 + bf[1]);
  hp[2] = tanhf(a2 + bf[2]);
}

// ---------------------------------------------------------------------------
// Masked instance-mean pooling
// ---------------------------------------------------------------------------
__global__ void zero_pool_kernel(float* __restrict__ acc, float* __restrict__ cnt) {
  int t = threadIdx.x;
  if (t < NB * 3) acc[t] = 0.f;
  if (t < NB)     cnt[t] = 0.f;
}

__global__ void pool_partial_kernel(const float* __restrict__ h, const int* __restrict__ inst,
                                    float* __restrict__ acc, float* __restrict__ cnt) {
  const int HWI = HH * WW;
  int b = blockIdx.y;
  int s = blockIdx.x;   // 64 slices
  float a0 = 0.f, a1 = 0.f, a2 = 0.f, cc = 0.f;
  for (int p = s * 256 + threadIdx.x; p < HWI; p += 64 * 256) {
    if (inst[(size_t)b * HWI + p] == 1) {
      const float* hp = h + ((size_t)b * HWI + p) * 3;
      a0 += hp[0]; a1 += hp[1]; a2 += hp[2]; cc += 1.f;
    }
  }
  __shared__ float sm[4][256];
  sm[0][threadIdx.x] = a0; sm[1][threadIdx.x] = a1;
  sm[2][threadIdx.x] = a2; sm[3][threadIdx.x] = cc;
  __syncthreads();
  for (int off = 128; off > 0; off >>= 1) {
    if (threadIdx.x < off) {
      #pragma unroll
      for (int q = 0; q < 4; ++q) sm[q][threadIdx.x] += sm[q][threadIdx.x + off];
    }
    __syncthreads();
  }
  if (threadIdx.x == 0) {
    atomicAdd(&acc[b * 3 + 0], sm[0][0]);
    atomicAdd(&acc[b * 3 + 1], sm[1][0]);
    atomicAdd(&acc[b * 3 + 2], sm[2][0]);
    atomicAdd(&cnt[b],         sm[3][0]);
  }
}

__global__ void scatter_kernel(const float* __restrict__ acc, const float* __restrict__ cnt,
                               const int* __restrict__ inst, float* __restrict__ out) {
  int id = blockIdx.x * blockDim.x + threadIdx.x;   // B*3*H*W (NCHW)
  int ox = id % WW;
  int t  = id / WW;
  int oy = t % HH;
  t /= HH;
  int c = t % 3;
  int b = t / 3;
  int m = inst[((size_t)b * HH + oy) * WW + ox];
  out[id] = (m == 1) ? acc[b * 3 + c] / cnt[b] : 0.f;
}

// ---------------------------------------------------------------------------
// Host driver
// ---------------------------------------------------------------------------
static inline void run_stats(const float* raw, float* psum, float* psq, float* mean, float* rstd,
                             int P, int C, hipStream_t stream) {
  stats_partial_kernel<<<dim3(NS, NB), 256, 0, stream>>>(raw, psum, psq, P, C);
  stats_combine_kernel<<<NB, 256, 0, stream>>>(psum, psq, mean, rstd, P, C);
}

static inline void launch_gemm(const _Float16* in, const _Float16* bp, float* out,
                               int Hp, int Wp, int C, int N, int Ho, int Wo, int stride,
                               hipStream_t stream) {
  dim3 grid(NB * Ho * Wo / 64, N / 64);
  dim3 blk(128);
  switch (C) {
    case 64:
      gemm_conv3x3_kernel<64><<<grid, blk, 0, stream>>>(in, bp, out, Hp, Wp, N, Ho, Wo, stride);
      break;
    case 128:
      gemm_conv3x3_kernel<128><<<grid, blk, 0, stream>>>(in, bp, out, Hp, Wp, N, Ho, Wo, stride);
      break;
    case 256:
      gemm_conv3x3_kernel<256><<<grid, blk, 0, stream>>>(in, bp, out, Hp, Wp, N, Ho, Wo, stride);
      break;
    case 512:
      gemm_conv3x3_kernel<512><<<grid, blk, 0, stream>>>(in, bp, out, Hp, Wp, N, Ho, Wo, stride);
      break;
    default:
      gemm_conv3x3_kernel<1024><<<grid, blk, 0, stream>>>(in, bp, out, Hp, Wp, N, Ho, Wo, stride);
      break;
  }
}

extern "C" void kernel_launch(void* const* d_in, const int* in_sizes, int n_in,
                              void* d_out, int out_size, void* d_ws, size_t ws_size,
                              hipStream_t stream) {
  (void)in_sizes; (void)n_in; (void)out_size; (void)ws_size;
  const float* x    = (const float*)d_in[0];
  const int*   inst = (const int*)d_in[1];
  const float* w0   = (const float*)d_in[2];
  const float* dw[4] = { (const float*)d_in[4],  (const float*)d_in[6],
                         (const float*)d_in[8],  (const float*)d_in[10] };
  const float* uw[4] = { (const float*)d_in[12], (const float*)d_in[14],
                         (const float*)d_in[16], (const float*)d_in[18] };
  const float* wf = (const float*)d_in[20];
  const float* bf = (const float*)d_in[21];
  float* out = (float*)d_out;

  char* ws = (char*)d_ws;
  float*    raw   = (float*)(ws + OFF_RAW);
  _Float16* padA  = (_Float16*)(ws + OFF_PADA);
  _Float16* padB  = (_Float16*)(ws + OFF_PADB);
  _Float16* bpack = (_Float16*)(ws + OFF_BP);
  float*    psum  = (float*)(ws + OFF_PSUM);
  float*    psq   = (float*)(ws + OFF_PSQ);
  float*    mean  = (float*)(ws + OFF_MEAN);
  float*    rstd  = (float*)(ws + OFF_RSTD);
  float*    pacc  = (float*)(ws + OFF_POOL);
  float*    pcnt  = pacc + NB * 3;

  // ---- stem: conv0 -> inorm -> relu -> pad1 f16 buffer ----
  {
    size_t nthr = (size_t)NB * HH * WW * 64;
    conv0_kernel<<<(unsigned)(nthr / 256), 256, 0, stream>>>(x, w0, raw);
    run_stats(raw, psum, psq, mean, rstd, HH * WW, 64, stream);
    size_t tot = (size_t)NB * 514 * 514 * 64;
    norm_kernel<<<(unsigned)((tot + 255) / 256), 256, 0, stream>>>(
        raw, mean, rstd, padA, HH, WW, 64, 514, 514, /*mode=*/0, tot);
  }

  _Float16* cur = padA;
  _Float16* nxt = padB;
  int Hin = 512, C = 64;

  // ---- downsampling stack: 4x (stride-2 3x3 conv via WMMA implicit GEMM) ----
  for (int i = 0; i < 4; ++i) {
    int Cout = C * 2;
    int Ho   = Hin / 2;
    int KN   = 9 * C * Cout;
    pack_down_kernel<<<(KN + 255) / 256, 256, 0, stream>>>(dw[i], bpack, C, Cout, KN);
    launch_gemm(cur, bpack, raw, Hin + 2, Hin + 2, C, Cout, Ho, Ho, /*stride=*/2, stream);
    run_stats(raw, psum, psq, mean, rstd, Ho * Ho, Cout, stream);
    int mode = (i < 3) ? 0 : 1;                    // last one feeds a transposed conv
    int Hp   = (i < 3) ? (Ho + 2) : (2 * Ho + 2);
    size_t tot = (size_t)NB * Hp * Hp * Cout;
    norm_kernel<<<(unsigned)((tot + 255) / 256), 256, 0, stream>>>(
        raw, mean, rstd, nxt, Ho, Ho, Cout, Hp, Hp, mode, tot);
    _Float16* t = cur; cur = nxt; nxt = t;
    Hin = Ho; C = Cout;
  }

  // ---- upsampling stack: 4x (convT == stride-1 conv on zero-dilated input) ----
  for (int i = 0; i < 4; ++i) {
    int Cout = C / 2;
    int Ho   = Hin * 2;
    int Hp   = 2 * Hin + 2;                        // dilated+padded input extent
    int KN   = 9 * C * Cout;
    pack_up_kernel<<<(KN + 255) / 256, 256, 0, stream>>>(uw[i], bpack, C, Cout, KN);
    launch_gemm(cur, bpack, raw, Hp, Hp, C, Cout, Ho, Ho, /*stride=*/1, stream);
    run_stats(raw, psum, psq, mean, rstd, Ho * Ho, Cout, stream);
    int mode = (i < 3) ? 1 : 2;
    int Hpn  = (i < 3) ? (2 * Ho + 2) : Ho;
    size_t tot = (size_t)NB * Hpn * Hpn * Cout;
    norm_kernel<<<(unsigned)((tot + 255) / 256), 256, 0, stream>>>(
        raw, mean, rstd, nxt, Ho, Ho, Cout, Hpn, Hpn, mode, tot);
    _Float16* t = cur; cur = nxt; nxt = t;
    Hin = Ho; C = Cout;
  }

  // ---- head: reflect 7x7 conv 64->3 + bias + tanh ----
  {
    size_t nthr = (size_t)NB * HH * WW;
    convf_kernel<<<(unsigned)(nthr / 256), 256, 0, stream>>>(cur, wf, bf, raw);
  }

  // ---- masked instance-mean pooling + scatter ----
  zero_pool_kernel<<<1, 32, 0, stream>>>(pacc, pcnt);
  pool_partial_kernel<<<dim3(64, NB), 256, 0, stream>>>(raw, inst, pacc, pcnt);
  {
    size_t nthr = (size_t)NB * 3 * HH * WW;
    scatter_kernel<<<(unsigned)(nthr / 256), 256, 0, stream>>>(pacc, pcnt, inst, out);
  }
}
